// ResNet_MoE_50113678409826
// MI455X (gfx1250) — matrix operand order repaired and checked by reading, not compiled, not run on hardware
//
#include <hip/hip_runtime.h>

// ============================================================================
// ResNet-18 (1ch) + MoE head for gfx1250 (MI455X), implicit-GEMM conv on
// V_WMMA_F32_16X16X32_F16.
// Block = 256 threads = 8 waves. Block tile = 64(M) x 128(N):
//   - B tile [128 N][32 K] staged in LDS via im2col gather (f32->f16);
//     each lane's B fragment is one contiguous 32-byte LDS read
//     (CDNA5 ISA B layout: lane L -> N=L%16, halves e = K=(e/8)*16+(L/16)*8+e%8).
//   - Weights pre-packed in *fragment order*: [kb][row][32 halves] where
//     halves f = lane_hi*16 + e store K = kb*32 + (e/8)*16 + lane_hi*8 + e%8,
//     so each A fragment is one contiguous 32-byte global load per lane.
//   - Each wave owns 16 N columns and accumulates FOUR 16x16 M tiles
//     (4 v_wmma per K step sharing one B fragment).
// ============================================================================

typedef __attribute__((ext_vector_type(16))) _Float16 v16h;
typedef __attribute__((ext_vector_type(8)))  float    v8f;

#define EPS_BN 1e-5f

// ---------------------------------------------------------------------------
// Prep kernels
// ---------------------------------------------------------------------------

// OIHW f32 weights -> fragment-ordered f16: dst[((kb*Cout)+row)*32 + f],
// f = lane_hi*16 + e  holds  K = kb*32 + (e/8)*16 + lane_hi*8 + (e%8).
__global__ void pack_w_kernel(const float* __restrict__ w, _Float16* __restrict__ dst,
                              int Cout, int Kreal, int Kpad, int n) {
  int i = blockIdx.x * 256 + threadIdx.x;
  if (i >= n) return;
  int f   = i & 31;
  int t   = i >> 5;
  int row = t % Cout;
  int kb  = t / Cout;
  int lane_hi = f >> 4;
  int e       = f & 15;
  int K = kb * 32 + ((e >> 3) << 4) + (lane_hi << 3) + (e & 7);
  dst[i] = (K < Kreal) ? (_Float16)w[row * Kreal + K] : (_Float16)0.0f;
}

// Fold BN into per-channel scale/shift.
__global__ void bn_fold_kernel(const float* __restrict__ g, const float* __restrict__ b,
                               const float* __restrict__ m, const float* __restrict__ vv,
                               float* __restrict__ sc, float* __restrict__ sh, int C) {
  int i = blockIdx.x * 64 + threadIdx.x;
  if (i >= C) return;
  float s = g[i] * rsqrtf(vv[i] + EPS_BN);
  sc[i] = s;
  sh[i] = b[i] - m[i] * s;
}

// ---------------------------------------------------------------------------
// Implicit-GEMM conv + BN (+ residual) (+ ReLU) on WMMA f16->f32.
//   GEMM: M = Cout, N = B*OH*OW, K = Cin*R*S (padded to 32).
// ---------------------------------------------------------------------------
template <int R, int S, int STRIDE, int PAD>
__global__ __launch_bounds__(256) void conv_wmma_kernel(
    const float* __restrict__ x, const _Float16* __restrict__ wf16,
    const float* __restrict__ scale, const float* __restrict__ shift,
    const float* __restrict__ residual, float* __restrict__ out,
    int B, int Cin, int H, int W, int Cout, int OH, int OW,
    int Kreal, int Kpad, int relu) {
  constexpr int RS = R * S;
  const int Ntot = B * OH * OW;
  const int n0   = blockIdx.x * 128;
  const int m0   = blockIdx.y * 64;   // 4 x 16-row M tiles per block

  const int tid     = threadIdx.x;
  const int lane    = tid & 31;
  const int wave    = tid >> 5;
  const int lane_lo = lane & 15;
  const int lane_hi = lane >> 4;

  // [128 N][32 K] f16 tile: each lane's B fragment is 16 contiguous halves.
  __shared__ __align__(64) _Float16 ldsB[128 * 32];

  // This thread's im2col column for staging: nl in [0,128), 16 K-halves.
  const int nl   = tid >> 1;
  const int koff = (tid & 1) * 16;
  const int n_g  = n0 + nl;
  int gb = 0, goh = 0, gow = 0;
  if (n_g < Ntot) {
    gb = n_g / (OH * OW);
    int r = n_g % (OH * OW);
    goh = r / OW;
    gow = r % OW;
  }
  const int ihbase = goh * STRIDE - PAD;
  const int iwbase = gow * STRIDE - PAD;

  v8f acc0 = {0.f, 0.f, 0.f, 0.f, 0.f, 0.f, 0.f, 0.f};
  v8f acc1 = acc0, acc2 = acc0, acc3 = acc0;

  // Fragment-ordered A: lane base for (kb, row) is ((kb/32)*Cout+row)*32
  // + lane_hi*16; consecutive lane_lo -> consecutive rows (coalesced).
  const _Float16* abase = wf16 + (size_t)(m0 + lane_lo) * 32 + lane_hi * 16;
  const size_t astep = (size_t)Cout * 32;     // advance one 32-wide K block

  for (int kb = 0; kb < Kpad; kb += 32) {
    // ---- stage B tile (im2col gather, f32 -> f16) ----
#pragma unroll
    for (int j = 0; j < 16; ++j) {
      int k = kb + ((j >> 3) << 4) + ((koff >> 4) << 3) + (j & 7);
      _Float16 hv = (_Float16)0.0f;
      if (n_g < Ntot && k < Kreal) {
        int ci = k / RS;
        int rs = k - ci * RS;
        int rr = rs / S;
        int ss = rs - rr * S;
        int ih = ihbase + rr;
        int iw = iwbase + ss;
        if (ih >= 0 && ih < H && iw >= 0 && iw < W)
          hv = (_Float16)x[(((size_t)gb * Cin + ci) * H + ih) * W + iw];
      }
      ldsB[nl * 32 + koff + j] = hv;
    }
    __syncthreads();

    // ---- A fragments: one 32-byte contiguous global load per tile ----
    const _Float16* ak = abase + (size_t)(kb >> 5) * astep;
    v16h a0 = *(const v16h*)(ak);
    v16h a1 = *(const v16h*)(ak + 16 * 32);
    v16h a2 = *(const v16h*)(ak + 32 * 32);
    v16h a3 = *(const v16h*)(ak + 48 * 32);

    // ---- B fragment: 32 contiguous bytes from LDS, shared by 4 WMMAs ----
    const int nlb = wave * 16 + lane_lo;
    v16h bfrag = *(const v16h*)&ldsB[nlb * 32 + lane_hi * 16];

    acc0 = __builtin_amdgcn_wmma_f32_16x16x32_f16(false, a0, false, bfrag, (short)0, acc0, false, false);
    acc1 = __builtin_amdgcn_wmma_f32_16x16x32_f16(false, a1, false, bfrag, (short)0, acc1, false, false);
    acc2 = __builtin_amdgcn_wmma_f32_16x16x32_f16(false, a2, false, bfrag, (short)0, acc2, false, false);
    acc3 = __builtin_amdgcn_wmma_f32_16x16x32_f16(false, a3, false, bfrag, (short)0, acc3, false, false);
    __syncthreads();
  }

  // ---- epilogue: D layout lane L -> N=L%16, M = vi + 8*(L/16) ----
  const int n_ep = n0 + wave * 16 + lane_lo;
  if (n_ep < Ntot) {
    int bb = n_ep / (OH * OW);
    int r2 = n_ep % (OH * OW);
    int oh2 = r2 / OW;
    int ow2 = r2 % OW;
    v8f accs[4] = {acc0, acc1, acc2, acc3};
#pragma unroll
    for (int t = 0; t < 4; ++t) {
#pragma unroll
      for (int vi = 0; vi < 8; ++vi) {
        int co = m0 + t * 16 + vi + 8 * lane_hi;
        float val = accs[t][vi] * scale[co] + shift[co];
        size_t oidx = (((size_t)bb * Cout + co) * OH + oh2) * OW + ow2;
        if (residual) val += residual[oidx];
        if (relu) val = val > 0.f ? val : 0.f;
        out[oidx] = val;
      }
    }
  }
}

// ---------------------------------------------------------------------------
// Pool / head kernels
// ---------------------------------------------------------------------------
__global__ void maxpool_kernel(const float* __restrict__ in, float* __restrict__ out,
                               int C, int H, int W, int OH, int OW, int total) {
  int idx = blockIdx.x * 256 + threadIdx.x;
  if (idx >= total) return;
  int ow = idx % OW;
  int t = idx / OW;
  int oh = t % OH; t /= OH;
  int c = t % C;
  int b = t / C;
  const float* p = in + ((size_t)b * C + c) * H * W;
  float mx = -3.402823466e38f;
#pragma unroll
  for (int kh = 0; kh < 3; ++kh) {
#pragma unroll
    for (int kw = 0; kw < 3; ++kw) {
      int ih = oh * 2 - 1 + kh;
      int iw = ow * 2 - 1 + kw;
      if (ih >= 0 && ih < H && iw >= 0 && iw < W) {
        float v = p[ih * W + iw];
        mx = v > mx ? v : mx;
      }
    }
  }
  out[idx] = mx;
}

__global__ void avgpool_kernel(const float* __restrict__ in, float* __restrict__ out,
                               int BC, int HW) {
  int idx = blockIdx.x * 256 + threadIdx.x;
  if (idx >= BC) return;
  const float* p = in + (size_t)idx * HW;
  float s = 0.f;
  for (int j = 0; j < HW; ++j) s += p[j];
  out[idx] = s / (float)HW;
}

__global__ void routing_kernel(const float* __restrict__ havg, const float* __restrict__ rw,
                               const float* __restrict__ rb, int* __restrict__ eid,
                               int B, int D, int E) {
  int b = blockIdx.x * 64 + threadIdx.x;
  if (b >= B) return;
  const float* h = havg + (size_t)b * D;
  float best = -3.402823466e38f;
  int bi = 0;
  for (int e = 0; e < E; ++e) {
    const float* w = rw + (size_t)e * D;
    float s = rb[e];
    for (int i = 0; i < D; ++i) s += h[i] * w[i];
    if (s > best) { best = s; bi = e; }  // strict '>' == first-max (jnp.argmax)
  }
  eid[b] = bi;
}

__global__ void expert_kernel(const float* __restrict__ havg, const float* __restrict__ fcw,
                              const float* __restrict__ fcb, const int* __restrict__ eid,
                              float* __restrict__ out, int D, int O) {
  int b = blockIdx.x;
  int o = threadIdx.x;
  if (o >= O) return;
  int e = eid[b];
  const float* h = havg + (size_t)b * D;
  const float* w = fcw + ((size_t)e * O + o) * D;
  float s = fcb[e * O + o];
  for (int i = 0; i < D; ++i) s += h[i] * w[i];
  out[b * O + o] = s;
}

// ---------------------------------------------------------------------------
// Host side
// ---------------------------------------------------------------------------
template <int R, int S, int STRIDE, int PAD>
static void run_conv(const float* x, const _Float16* wf16, const float* sc, const float* sh,
                     const float* res, float* out, int B, int Cin, int H, int W, int Cout,
                     int relu, hipStream_t stream) {
  int OH = (H + 2 * PAD - R) / STRIDE + 1;
  int OW = (W + 2 * PAD - S) / STRIDE + 1;
  int Kreal = Cin * R * S;
  int Kpad  = (Kreal + 31) & ~31;
  long Ntot = (long)B * OH * OW;
  dim3 grid((unsigned)((Ntot + 127) / 128), (unsigned)(Cout / 64), 1);
  conv_wmma_kernel<R, S, STRIDE, PAD><<<grid, 256, 0, stream>>>(
      x, wf16, sc, sh, res, out, B, Cin, H, W, Cout, OH, OW, Kreal, Kpad, relu);
}

struct BnPtr { const float *g, *b, *m, *v; };
struct Blk   { const float* w1; BnPtr bn1; const float* w2; BnPtr bn2;
               const float* wd; BnPtr bnd; bool down; };
struct PrepOut { const _Float16* w; const float* sc; const float* sh; };

extern "C" void kernel_launch(void* const* d_in, const int* in_sizes, int n_in,
                              void* d_out, int out_size, void* d_ws, size_t ws_size,
                              hipStream_t stream) {
  (void)in_sizes; (void)n_in; (void)out_size; (void)ws_size;

  // ---- input pointer walk (Python dict insertion order, recursive) ----
  int ii = 0;
  auto IN = [&](void) { return (const float*)d_in[ii++]; };
  const float* x      = IN();                 // x: [64,1,224,224]
  const float* w_stem = IN();                 // conv1_w [64,1,7,7]
  BnPtr bn_stem; bn_stem.g = IN(); bn_stem.b = IN(); bn_stem.m = IN(); bn_stem.v = IN();
  Blk blk[4][2];
  for (int si = 0; si < 4; ++si)
    for (int bi = 0; bi < 2; ++bi) {
      Blk& B_ = blk[si][bi];
      B_.w1 = IN(); B_.bn1.g = IN(); B_.bn1.b = IN(); B_.bn1.m = IN(); B_.bn1.v = IN();
      B_.w2 = IN(); B_.bn2.g = IN(); B_.bn2.b = IN(); B_.bn2.m = IN(); B_.bn2.v = IN();
      B_.down = (si > 0 && bi == 0);
      if (B_.down) {
        B_.wd = IN(); B_.bnd.g = IN(); B_.bnd.b = IN(); B_.bnd.m = IN(); B_.bnd.v = IN();
      } else {
        B_.wd = nullptr; B_.bnd = BnPtr{nullptr, nullptr, nullptr, nullptr};
      }
    }
  const float* routing_w = IN();
  const float* routing_b = IN();
  const float* fc_w      = IN();
  const float* fc_b      = IN();

  // ---- workspace carving ----
  char* ws = (char*)d_ws;
  size_t off = 0;
  auto carve = [&](size_t bytes) -> void* {
    off = (off + 255) & ~(size_t)255;
    void* p = ws + off;
    off += bytes;
    return p;
  };
  const int B = 64;
  float*     stemOut = (float*)carve((size_t)64 * 64 * 112 * 112 * 4);   // 205 MB
  float*     bufs[4];
  for (int j = 0; j < 4; ++j)
    bufs[j] = (float*)carve((size_t)64 * 64 * 56 * 56 * 4);              // 51 MB each
  _Float16*  wbuf  = (_Float16*)carve((size_t)16 * 1024 * 1024 * 2);     // 32 MB f16 weights
  float*     bnbuf = (float*)carve((size_t)32768 * 4);
  float*     havg  = (float*)carve((size_t)64 * 512 * 4);
  int*       eid   = (int*)carve((size_t)64 * 4);

  size_t wcur = 0, bncur = 0;
  auto prep = [&](const float* w, BnPtr bn, int Cout, int Kreal) -> PrepOut {
    int Kpad = (Kreal + 31) & ~31;
    _Float16* wp = wbuf + wcur;
    wcur += (size_t)Cout * Kpad;
    int n = Cout * Kpad;
    pack_w_kernel<<<(n + 255) / 256, 256, 0, stream>>>(w, wp, Cout, Kreal, Kpad, n);
    float* sc = bnbuf + bncur; bncur += Cout;
    float* sh = bnbuf + bncur; bncur += Cout;
    bn_fold_kernel<<<(Cout + 63) / 64, 64, 0, stream>>>(bn.g, bn.b, bn.m, bn.v, sc, sh, Cout);
    return PrepOut{wp, sc, sh};
  };

  // ---- stem: conv 7x7 s2 p3 + BN + ReLU, then maxpool 3x3 s2 p1 ----
  PrepOut ps = prep(w_stem, bn_stem, 64, 1 * 7 * 7);
  run_conv<7, 7, 2, 3>(x, ps.w, ps.sc, ps.sh, nullptr, stemOut, B, 1, 224, 224, 64, 1, stream);
  {
    int total = B * 64 * 56 * 56;
    maxpool_kernel<<<(total + 255) / 256, 256, 0, stream>>>(stemOut, bufs[0], 64, 112, 112, 56, 56, total);
  }

  // ---- 4 stages x 2 basic blocks ----
  int hcur = 0;
  int H = 56;
  int cin = 64;
  for (int si = 0; si < 4; ++si) {
    int cout = 64 << si;
    for (int bi = 0; bi < 2; ++bi) {
      Blk& bk = blk[si][bi];
      int stride = (si > 0 && bi == 0) ? 2 : 1;
      int cinB   = (bi == 0) ? cin : cout;
      int Hout   = H / stride;

      // pick three scratch buffers != current h
      int others[3], oc = 0;
      for (int j = 0; j < 4; ++j) if (j != hcur) others[oc++] = j;
      float* tTmp  = bufs[others[0]];
      float* tDown = bufs[others[1]];
      float* tOut  = bufs[others[2]];

      PrepOut p1 = prep(bk.w1, bk.bn1, cout, cinB * 9);
      if (stride == 1)
        run_conv<3, 3, 1, 1>(bufs[hcur], p1.w, p1.sc, p1.sh, nullptr, tTmp, B, cinB, H, H, cout, 1, stream);
      else
        run_conv<3, 3, 2, 1>(bufs[hcur], p1.w, p1.sc, p1.sh, nullptr, tTmp, B, cinB, H, H, cout, 1, stream);

      const float* res;
      if (bk.down) {
        PrepOut pd = prep(bk.wd, bk.bnd, cout, cinB);
        run_conv<1, 1, 2, 0>(bufs[hcur], pd.w, pd.sc, pd.sh, nullptr, tDown, B, cinB, H, H, cout, 0, stream);
        res = tDown;
      } else {
        res = bufs[hcur];
      }

      PrepOut p2 = prep(bk.w2, bk.bn2, cout, cout * 9);
      run_conv<3, 3, 1, 1>(tTmp, p2.w, p2.sc, p2.sh, res, tOut, B, cout, Hout, Hout, cout, 1, stream);

      hcur = others[2];
      H = Hout;
    }
    cin = cout;
  }

  // ---- head: avgpool -> routing argmax -> per-sample expert GEMV ----
  {
    int BC = B * 512;
    avgpool_kernel<<<(BC + 255) / 256, 256, 0, stream>>>(bufs[hcur], havg, BC, 7 * 7);
    routing_kernel<<<1, 64, 0, stream>>>(havg, routing_w, routing_b, eid, B, 512, 4);
    expert_kernel<<<B, 16, 0, stream>>>(havg, fc_w, fc_b, eid, (float*)d_out, 512, 9);
  }
}
